// GATv2_33578054320559
// MI455X (gfx1250) — compile-verified, matrix operand order checked by reading, not run on hardware
//
#include <hip/hip_runtime.h>
#include <math.h>
#include <stdint.h>

#define NN 100000      // nodes
#define NE 400000      // edges
#define NEA 500000     // edges + self loops
#define NG 4000        // graphs
#define INCH 64
#define EDIM 16
#define HC 128         // heads*out_ch
#define CATD 640
#define CLSD 512

typedef __attribute__((ext_vector_type(16))) _Float16 v16h;
typedef __attribute__((ext_vector_type(8)))  _Float16 v8h;
typedef __attribute__((ext_vector_type(8)))  float    v8f;

#if defined(__has_builtin)
#  if __has_builtin(__builtin_amdgcn_sched_barrier)
#    define SCHED_BARRIER() __builtin_amdgcn_sched_barrier(0)
#  endif
#endif
#ifndef SCHED_BARRIER
#  define SCHED_BARRIER() asm volatile("" ::: "memory")
#endif

// ---------------------------------------------------------------- utilities
__global__ void fill_f32(float* p, float v, int n) {
  int i = blockIdx.x * blockDim.x + threadIdx.x;
  if (i < n) p[i] = v;
}
__global__ void fill_u32(unsigned* p, unsigned v, int n) {
  int i = blockIdx.x * blockDim.x + threadIdx.x;
  if (i < n) p[i] = v;
}

__device__ __forceinline__ unsigned enc_f32(float f) {
  unsigned b = __float_as_uint(f);
  return (b & 0x80000000u) ? ~b : (b | 0x80000000u);
}
__device__ __forceinline__ float dec_f32(unsigned u) {
  unsigned b = (u & 0x80000000u) ? (u & 0x7FFFFFFFu) : ~u;
  return __uint_as_float(b);
}

// ---------------------------------------------------- f32 -> f16 staging
__global__ void cvt_f2h(const float* __restrict__ src, _Float16* __restrict__ dst, int n) {
  int i = blockIdx.x * blockDim.x + threadIdx.x;
  if (i < n) dst[i] = (_Float16)src[i];
}
// strided activation copy: one block per row, dst packed rows x K
__global__ void cvt_rows_f2h(const float* __restrict__ src, int lda,
                             _Float16* __restrict__ dst, int K) {
  const int row = blockIdx.x;
  for (int c = threadIdx.x; c < K; c += blockDim.x)
    dst[(size_t)row * K + c] = (_Float16)src[(size_t)row * lda + c];
}

__device__ __forceinline__ v16h cat16(v8h lo, v8h hi) {
  return __builtin_shufflevector(lo, hi, 0, 1, 2, 3, 4, 5, 6, 7,
                                 8, 9, 10, 11, 12, 13, 14, 15);
}

// ------------------------------------------------- WMMA f16 GEMM (f32 accum)
// C[n, colBase..colBase+16*NT) = act( A[n,:K] @ W[col,:K]^T + bias[col] )
// A: nrows x K packed f16, W: M x K packed f16. K % 32 == 0.
// NT=4 keeps acc(32)+B(32)+A(8) VGPRs spill-free while the sched barrier
// forces one batched load clause per K-step, drained by 4 streamed WMMAs.
template <int NT>
__global__ void gemm_wmma(const _Float16* __restrict__ A,
                          const _Float16* __restrict__ W,
                          const float* __restrict__ bias,
                          float* __restrict__ C, int ldc,
                          int nrows, int K, float slope) {
  const int lane = threadIdx.x & 31;
  const int tileRow = blockIdx.x * 16;
  const int colBase = blockIdx.y * (16 * NT);
  int rowA = tileRow + (lane & 15);
  if (rowA >= nrows) rowA = nrows - 1;
  const int koff = (lane >> 4) * 8;   // lanes 16-31 hold the K+8 chunk (ISA 7.12.2)

  v8f acc[NT];
#pragma unroll
  for (int j = 0; j < NT; ++j) acc[j] = v8f{0.f,0.f,0.f,0.f,0.f,0.f,0.f,0.f};

  const _Float16* arow = A + (size_t)rowA * K + koff;
  const _Float16* wcol = W + (size_t)(colBase + (lane & 15)) * K + koff;

  for (int k0 = 0; k0 < K; k0 += 32) {
    // ---- issue all loads for this K-step (one clause)
    v16h a = cat16(*(const v8h*)(arow + k0),        // K = k0+koff+[0,8)
                   *(const v8h*)(arow + k0 + 16));  // K = k0+koff+[16,24)
    v16h b[NT];
#pragma unroll
    for (int j = 0; j < NT; ++j) {
      const _Float16* wp = wcol + (size_t)j * 16 * K + k0;
      b[j] = cat16(*(const v8h*)(wp), *(const v8h*)(wp + 16));
    }
    SCHED_BARRIER();   // keep fragments live -> distinct regs, batched loads
    // ---- stream the WMMAs (distinct acc/B regs: no hazard NOPs between)
#pragma unroll
    for (int j = 0; j < NT; ++j) {
      acc[j] = __builtin_amdgcn_wmma_f32_16x16x32_f16(
          false, a, false, b[j], (short)0, acc[j], false, false);
    }
  }

  const int rbase = tileRow + ((lane >> 4) * 8);  // C layout: lanes 16-31 -> M+8
  const int cl = lane & 15;
#pragma unroll
  for (int j = 0; j < NT; ++j) {
    const int col = colBase + j * 16 + cl;
    const float bv = bias ? bias[col] : 0.0f;
#pragma unroll
    for (int r = 0; r < 8; ++r) {
      const int row = rbase + r;
      if (row < nrows) {
        float v = acc[j][r] + bv;
        if (slope != 1.0f) v = v > 0.f ? v : v * slope;
        C[(size_t)row * ldc + col] = v;
      }
    }
  }
}

// ------------------------------------------------------------- graph setup
__global__ void copy_edge_attr(const float* __restrict__ ea, float* __restrict__ dst) {
  int i = blockIdx.x * blockDim.x + threadIdx.x;
  if (i < NE * EDIM) dst[i] = ea[i];
}
__global__ void build_idx(const int* __restrict__ eidx, int* __restrict__ src_a,
                          int* __restrict__ dst_a) {
  int i = blockIdx.x * blockDim.x + threadIdx.x;
  if (i < NEA) {
    src_a[i] = (i < NE) ? eidx[i] : (i - NE);
    dst_a[i] = (i < NE) ? eidx[NE + i] : (i - NE);
  }
}
__global__ void loop_accum(const float* __restrict__ ea, const int* __restrict__ eidx,
                           float* __restrict__ loop_attr, float* __restrict__ counts) {
  int e = blockIdx.x * blockDim.x + threadIdx.x;
  if (e < NE) {
    int d = eidx[NE + e];
    atomicAdd(&counts[d], 1.0f);
    for (int j = 0; j < EDIM; ++j)
      atomicAdd(&loop_attr[(size_t)d * EDIM + j], ea[(size_t)e * EDIM + j]);
  }
}
__global__ void loop_div(float* __restrict__ loop_attr, const float* __restrict__ counts) {
  int i = blockIdx.x * blockDim.x + threadIdx.x;
  if (i < NN * EDIM) loop_attr[i] /= fmaxf(counts[i >> 4], 1.0f);
}

// --------------------------------------------------------------- GAT edges
// wave per edge: lane handles channels c = lane + 32*h, h = 0..3
__global__ void edge_logits(const float* __restrict__ xl, const float* __restrict__ xr,
                            const float* __restrict__ eattr, const int* __restrict__ src,
                            const int* __restrict__ dst, const float* __restrict__ We,
                            const float* __restrict__ att, float* __restrict__ logits,
                            unsigned* __restrict__ lmaxu, int nE) {
  __shared__ float sWe[HC * EDIM];
  __shared__ float sAtt[HC];
  for (int i = threadIdx.x; i < HC * EDIM; i += blockDim.x) sWe[i] = We[i];
  if (threadIdx.x < HC) sAtt[threadIdx.x] = att[threadIdx.x];
  __syncthreads();

  const int lane = threadIdx.x & 31;
  const int e = blockIdx.x * 8 + (threadIdx.x >> 5);
  float p[4] = {0.f, 0.f, 0.f, 0.f};
  int d = 0;
  if (e < nE) {
    const int s = src[e];
    d = dst[e];
    float ev[EDIM];
#pragma unroll
    for (int j = 0; j < EDIM; ++j) ev[j] = eattr[(size_t)e * EDIM + j];
#pragma unroll
    for (int h = 0; h < 4; ++h) {
      const int c = lane + 32 * h;
      float ec = 0.f;
#pragma unroll
      for (int j = 0; j < EDIM; ++j) ec += ev[j] * sWe[c * EDIM + j];
      float m = xl[(size_t)s * HC + c] + xr[(size_t)d * HC + c] + ec;
      m = m > 0.f ? m : 0.2f * m;               // GATv2 negative_slope
      p[h] = m * sAtt[c];
    }
  }
#pragma unroll
  for (int h = 0; h < 4; ++h)
    for (int off = 16; off > 0; off >>= 1) p[h] += __shfl_xor(p[h], off, 32);
  if (e < nE && lane == 0) {
#pragma unroll
    for (int h = 0; h < 4; ++h) {
      logits[(size_t)e * 4 + h] = p[h];
      atomicMax(&lmaxu[(size_t)d * 4 + h], enc_f32(p[h]));
    }
  }
}

__global__ void edge_exp(float* __restrict__ logits, const unsigned* __restrict__ lmaxu,
                         float* __restrict__ denom, const int* __restrict__ dst, int nE) {
  int i = blockIdx.x * blockDim.x + threadIdx.x;
  if (i < nE * 4) {
    int e = i >> 2, h = i & 3;
    int d = dst[e];
    float ex = expf(logits[i] - dec_f32(lmaxu[(size_t)d * 4 + h]));
    logits[i] = ex;                              // overwrite in place
    atomicAdd(&denom[(size_t)d * 4 + h], ex);
  }
}

__global__ void edge_aggr(const float* __restrict__ xl, const float* __restrict__ ex,
                          const float* __restrict__ denom, const int* __restrict__ src,
                          const int* __restrict__ dst, float* __restrict__ cat,
                          int ofs, int nE) {
  const int lane = threadIdx.x & 31;
  const int e = blockIdx.x * 8 + (threadIdx.x >> 5);
  if (e >= nE) return;
  const int s = src[e], d = dst[e];
#pragma unroll
  for (int h = 0; h < 4; ++h) {
    const int c = lane + 32 * h;
    const float alpha = ex[(size_t)e * 4 + h] / denom[(size_t)d * 4 + h];
    atomicAdd(&cat[(size_t)d * CATD + ofs + c], xl[(size_t)s * HC + c] * alpha);
  }
}

__global__ void zero_cat_cols(float* __restrict__ cat, int ofs) {
  int i = blockIdx.x * blockDim.x + threadIdx.x;
  if (i < NN * HC) cat[(size_t)(i >> 7) * CATD + ofs + (i & 127)] = 0.0f;
}
__global__ void bias_act(float* __restrict__ cat, int ofs, const float* __restrict__ bias) {
  int i = blockIdx.x * blockDim.x + threadIdx.x;
  if (i < NN * HC) {
    int n = i >> 7, c = i & 127;
    float v = cat[(size_t)n * CATD + ofs + c] + bias[c];
    cat[(size_t)n * CATD + ofs + c] = v > 0.f ? v : 0.01f * v;  // outer LeakyReLU
  }
}

// ----------------------------------------------------------------- pooling
__global__ void pool_accum(const float* __restrict__ cat, const int* __restrict__ batch,
                           float* __restrict__ hpool) {
  int i = blockIdx.x * blockDim.x + threadIdx.x;
  if (i < NN * HC) {
    int n = i >> 7, c = i & 127;
    atomicAdd(&hpool[(size_t)batch[n] * HC + c], cat[(size_t)n * CATD + 384 + c]);
  }
}
__global__ void pool_bcast(float* __restrict__ cat, const int* __restrict__ batch,
                           const float* __restrict__ hpool) {
  int i = blockIdx.x * blockDim.x + threadIdx.x;
  if (i < NN * HC) {
    int n = i >> 7, c = i & 127;
    cat[(size_t)n * CATD + 512 + c] = hpool[(size_t)batch[n] * HC + c];
  }
}

// --------------------------------------------------------------- final head
__global__ void final_sig(const float* __restrict__ buf, const float* __restrict__ finW,
                          const float* __restrict__ finb, float* __restrict__ out) {
  const int lane = threadIdx.x & 31;
  const int node = blockIdx.x * 8 + (threadIdx.x >> 5);
  float acc = 0.f;
  if (node < NN) {
#pragma unroll
    for (int i = 0; i < CLSD / 32; ++i) {
      int c = lane + 32 * i;
      acc += buf[(size_t)node * CLSD + c] * finW[c];
    }
  }
  for (int off = 16; off > 0; off >>= 1) acc += __shfl_xor(acc, off, 32);
  if (node < NN && lane == 0) out[node] = 1.0f / (1.0f + expf(-(acc + finb[0])));
}

// ------------------------------------------------------------------- driver
static inline int cdiv(int a, int b) { return (a + b - 1) / b; }
static inline char* alignp(char* p, size_t a) {
  return (char*)(((uintptr_t)p + a - 1) & ~(uintptr_t)(a - 1));
}

struct Scratch {
  float* eattr_a; int* src_a; int* dst_a; float* counts;
  float* xl; float* xr; float* logits; unsigned* lmaxu; float* denom;
  float* cat; float* hpool; float* buf1; float* buf2;
  _Float16* ah;                   // staged f16 activations (NN x CATD max)
  _Float16* wh_c1l; _Float16* wh_c1r;
  _Float16* wh_csl; _Float16* wh_csr;   // 3 layers each
  _Float16* wh_cls1; _Float16* wh_cls;  // 2 layers
};

static void gat_layer(const float* hin, int lda, int K,
                      const _Float16* Wlh, const float* bl,
                      const _Float16* Wrh, const float* br,
                      const float* We, const float* att, const float* bias,
                      int outOfs, const Scratch& S, hipStream_t stream) {
  // stage activations once, reuse for both transforms
  cvt_rows_f2h<<<NN, 256, 0, stream>>>(hin, lda, S.ah, K);
  dim3 gg(NN / 16, HC / 64);
  gemm_wmma<4><<<gg, 32, 0, stream>>>(S.ah, Wlh, bl, S.xl, HC, NN, K, 1.0f);
  gemm_wmma<4><<<gg, 32, 0, stream>>>(S.ah, Wrh, br, S.xr, HC, NN, K, 1.0f);
  fill_u32<<<cdiv(NN * 4, 256), 256, 0, stream>>>(S.lmaxu, 0u, NN * 4);
  fill_f32<<<cdiv(NN * 4, 256), 256, 0, stream>>>(S.denom, 0.0f, NN * 4);
  zero_cat_cols<<<cdiv(NN * HC, 256), 256, 0, stream>>>(S.cat, outOfs);
  edge_logits<<<cdiv(NEA, 8), 256, 0, stream>>>(S.xl, S.xr, S.eattr_a, S.src_a, S.dst_a,
                                                We, att, S.logits, S.lmaxu, NEA);
  edge_exp<<<cdiv(NEA * 4, 256), 256, 0, stream>>>(S.logits, S.lmaxu, S.denom, S.dst_a, NEA);
  edge_aggr<<<cdiv(NEA, 8), 256, 0, stream>>>(S.xl, S.logits, S.denom, S.src_a, S.dst_a,
                                              S.cat, outOfs, NEA);
  bias_act<<<cdiv(NN * HC, 256), 256, 0, stream>>>(S.cat, outOfs, bias);
}

extern "C" void kernel_launch(void* const* d_in, const int* in_sizes, int n_in,
                              void* d_out, int out_size, void* d_ws, size_t ws_size,
                              hipStream_t stream) {
  const float* x         = (const float*)d_in[0];
  const float* edge_attr = (const float*)d_in[1];
  const float* c1_Wl = (const float*)d_in[2];  const float* c1_bl = (const float*)d_in[3];
  const float* c1_Wr = (const float*)d_in[4];  const float* c1_br = (const float*)d_in[5];
  const float* c1_We = (const float*)d_in[6];  const float* c1_att = (const float*)d_in[7];
  const float* c1_bias = (const float*)d_in[8];
  const float* cs_Wl = (const float*)d_in[9];  const float* cs_bl = (const float*)d_in[10];
  const float* cs_Wr = (const float*)d_in[11]; const float* cs_br = (const float*)d_in[12];
  const float* cs_We = (const float*)d_in[13]; const float* cs_att = (const float*)d_in[14];
  const float* cs_bias = (const float*)d_in[15];
  const float* cls1_W = (const float*)d_in[16]; const float* cls1_b = (const float*)d_in[17];
  const float* cls_W = (const float*)d_in[18];  const float* cls_b = (const float*)d_in[19];
  const float* fin_W = (const float*)d_in[20];  const float* fin_b = (const float*)d_in[21];
  const int* edge_index = (const int*)d_in[22];
  const int* batch = (const int*)d_in[23];
  float* out = (float*)d_out;

  // ---- carve workspace (keep 64B alignment for vector loads)
  char* pc = (char*)d_ws;
  Scratch S;
  auto takeF = [&](size_t n) { pc = alignp(pc, 64); float* r = (float*)pc; pc += n * 4; return r; };
  auto takeI = [&](size_t n) { pc = alignp(pc, 64); int* r = (int*)pc; pc += n * 4; return r; };
  auto takeH = [&](size_t n) { pc = alignp(pc, 64); _Float16* r = (_Float16*)pc; pc += n * 2; return r; };
  S.eattr_a = takeF((size_t)NEA * EDIM);
  S.src_a = takeI(NEA);
  S.dst_a = takeI(NEA);
  S.counts = takeF(NN);
  S.xl = takeF((size_t)NN * HC);
  S.xr = takeF((size_t)NN * HC);
  S.logits = takeF((size_t)NEA * 4);
  S.lmaxu = (unsigned*)takeI((size_t)NN * 4);
  S.denom = takeF((size_t)NN * 4);
  S.cat = takeF((size_t)NN * CATD);
  S.hpool = takeF((size_t)NG * HC);
  S.buf1 = takeF((size_t)NN * CLSD);
  S.buf2 = takeF((size_t)NN * CLSD);
  S.ah = takeH((size_t)NN * CATD);
  S.wh_c1l = takeH((size_t)HC * INCH);
  S.wh_c1r = takeH((size_t)HC * INCH);
  S.wh_csl = takeH((size_t)3 * HC * HC);
  S.wh_csr = takeH((size_t)3 * HC * HC);
  S.wh_cls1 = takeH((size_t)CLSD * CATD);
  S.wh_cls = takeH((size_t)2 * CLSD * CLSD);

  // ---- stage all weights to f16 (tiny, once per launch)
  cvt_f2h<<<cdiv(HC * INCH, 256), 256, 0, stream>>>(c1_Wl, S.wh_c1l, HC * INCH);
  cvt_f2h<<<cdiv(HC * INCH, 256), 256, 0, stream>>>(c1_Wr, S.wh_c1r, HC * INCH);
  cvt_f2h<<<cdiv(3 * HC * HC, 256), 256, 0, stream>>>(cs_Wl, S.wh_csl, 3 * HC * HC);
  cvt_f2h<<<cdiv(3 * HC * HC, 256), 256, 0, stream>>>(cs_Wr, S.wh_csr, 3 * HC * HC);
  cvt_f2h<<<cdiv(CLSD * CATD, 256), 256, 0, stream>>>(cls1_W, S.wh_cls1, CLSD * CATD);
  cvt_f2h<<<cdiv(2 * CLSD * CLSD, 256), 256, 0, stream>>>(cls_W, S.wh_cls, 2 * CLSD * CLSD);

  // ---- self-loop edge attrs (fill_value='mean') + augmented edge lists
  float* loop_attr = S.eattr_a + (size_t)NE * EDIM;
  fill_f32<<<cdiv(NN * EDIM, 256), 256, 0, stream>>>(loop_attr, 0.0f, NN * EDIM);
  fill_f32<<<cdiv(NN, 256), 256, 0, stream>>>(S.counts, 0.0f, NN);
  copy_edge_attr<<<cdiv(NE * EDIM, 256), 256, 0, stream>>>(edge_attr, S.eattr_a);
  build_idx<<<cdiv(NEA, 256), 256, 0, stream>>>(edge_index, S.src_a, S.dst_a);
  loop_accum<<<cdiv(NE, 256), 256, 0, stream>>>(edge_attr, edge_index, loop_attr, S.counts);
  loop_div<<<cdiv(NN * EDIM, 256), 256, 0, stream>>>(loop_attr, S.counts);

  // ---- GAT stack; layer L output occupies cat columns [L*128, L*128+128)
  gat_layer(x, INCH, INCH, S.wh_c1l, c1_bl, S.wh_c1r, c1_br, c1_We, c1_att, c1_bias,
            0, S, stream);
  for (int i = 0; i < 3; ++i) {
    gat_layer(S.cat + i * HC, CATD, HC,
              S.wh_csl + (size_t)i * HC * HC, cs_bl + (size_t)i * HC,
              S.wh_csr + (size_t)i * HC * HC, cs_br + (size_t)i * HC,
              cs_We + (size_t)i * HC * EDIM, cs_att + (size_t)i * HC,
              cs_bias + (size_t)i * HC, (i + 1) * HC, S, stream);
  }

  // ---- global add pool, broadcast back into cat cols [512,640)
  fill_f32<<<cdiv(NG * HC, 256), 256, 0, stream>>>(S.hpool, 0.0f, NG * HC);
  pool_accum<<<cdiv(NN * HC, 256), 256, 0, stream>>>(S.cat, batch, S.hpool);
  pool_bcast<<<cdiv(NN * HC, 256), 256, 0, stream>>>(S.cat, batch, S.hpool);

  // ---- classifier MLP (WMMA GEMMs, f16-staged inputs)
  dim3 gc(NN / 16, CLSD / 64);
  cvt_rows_f2h<<<NN, 256, 0, stream>>>(S.cat, CATD, S.ah, CATD);
  gemm_wmma<4><<<gc, 32, 0, stream>>>(S.ah, S.wh_cls1, cls1_b,
                                      S.buf1, CLSD, NN, CATD, 1.0f);      // no activation
  cvt_rows_f2h<<<NN, 256, 0, stream>>>(S.buf1, CLSD, S.ah, CLSD);
  gemm_wmma<4><<<gc, 32, 0, stream>>>(S.ah, S.wh_cls, cls_b,
                                      S.buf2, CLSD, NN, CLSD, 0.01f);
  cvt_rows_f2h<<<NN, 256, 0, stream>>>(S.buf2, CLSD, S.ah, CLSD);
  gemm_wmma<4><<<gc, 32, 0, stream>>>(S.ah, S.wh_cls + (size_t)CLSD * CLSD,
                                      cls_b + CLSD, S.buf1, CLSD, NN, CLSD, 0.01f);
  final_sig<<<cdiv(NN, 8), 256, 0, stream>>>(S.buf1, fin_W, fin_b, out);
}
